// CAGERF_GNN_72765335928998
// MI455X (gfx1250) — matrix operand order repaired and checked by reading, not compiled, not run on hardware
//
#include <hip/hip_runtime.h>
#include <hip/hip_bf16.h>

typedef __attribute__((ext_vector_type(16))) _Float16 v16h;
typedef __attribute__((ext_vector_type(8)))  float    v8f;
typedef __attribute__((ext_vector_type(4)))  float    f4;

#define NN  100000
#define EE  1600000
#define RR  6
#define DIN 128
#define HH  64

// ---------------- atomics: hardware f32 add, relaxed/agent ----------------
__device__ __forceinline__ void atomAddF32(float* p, float v) {
    __hip_atomic_fetch_add(p, v, __ATOMIC_RELAXED, __HIP_MEMORY_SCOPE_AGENT);
}

// ---------------- graph normalization ----------------
__global__ void deg_kernel(const int* __restrict__ src, float* __restrict__ deg) {
    int i = blockIdx.x * blockDim.x + threadIdx.x;
    if (i < EE) atomAddF32(&deg[src[i]], 1.0f);
}

__global__ void dis_kernel(float* __restrict__ deg) {
    int i = blockIdx.x * blockDim.x + threadIdx.x;
    if (i < NN) {
        float d = deg[i];
        deg[i] = (d > 0.0f) ? rsqrtf(d) : 0.0f;
    }
}

__global__ void norm_kernel(const int* __restrict__ src, const int* __restrict__ dst,
                            const float* __restrict__ dis, float* __restrict__ nrm) {
    int i = blockIdx.x * blockDim.x + threadIdx.x;
    if (i < EE) nrm[i] = -(dis[src[i]] * dis[dst[i]]);
}

__global__ void negcopy_kernel(const float* __restrict__ in, float* __restrict__ out, int n) {
    int i = blockIdx.x * blockDim.x + threadIdx.x;
    if (i < n) out[i] = -in[i];
}

// ---------------- SpMM scatter: vout[dst] += scale*norm[e]*vin[src] ----------------
// One thread per (edge, 4-feature chunk); a wave covers consecutive chunks of one
// (or two) edge rows -> coalesced float4 gathers and coalesced atomic adds.
__global__ void spmm_kernel(const int* __restrict__ src, const int* __restrict__ dst,
                            const float* __restrict__ nrm,
                            const float* __restrict__ vin, float* __restrict__ vout,
                            int d4, float scale) {
    long long t = (long long)blockIdx.x * blockDim.x + threadIdx.x;
    int e = (int)(t / d4);
    int c = (int)(t % d4);
    if (e >= EE) return;
    int D = d4 * 4;
    const f4 v = ((const f4*)(vin + (size_t)src[e] * D))[c];
    float s = scale * nrm[e];
    float* base = vout + (size_t)dst[e] * D + (size_t)c * 4;
    atomAddF32(base + 0, s * v.x);
    atomAddF32(base + 1, s * v.y);
    atomAddF32(base + 2, s * v.z);
    atomAddF32(base + 3, s * v.w);
}

// ---------------- fused Chebyshev GEMM: out = relu(A0@W0 + A1@W1 + A2@W2 + b) ------
// WMMA f32_16x16x32_f16. Block = 128 threads = 4 waves; wave w owns the 16-col tile
// w of a 16-row stripe. grid.x = NN/16 (exact).
__global__ __launch_bounds__(128)
void cheb_gemm_kernel(const float* __restrict__ A0, const float* __restrict__ A1,
                      const float* __restrict__ A2, const float* __restrict__ W,
                      const float* __restrict__ bias, float* __restrict__ out,
                      int din, int ostride, int ooff, int do_relu) {
    const int lane = threadIdx.x & 31;
    const int ct   = threadIdx.x >> 5;     // 0..3 : 16-column tile
    const int row  = lane & 15;
    const int hi   = lane >> 4;            // 0/1 half-wave
    const int m    = blockIdx.x * 16 + row;       // A row for this lane
    const int col  = ct * 16 + row;               // B/C column for this lane
    const float* As[3] = {A0, A1, A2};

    v8f c = {};
    for (int t = 0; t < 3; ++t) {
        const float* Abase = As[t] + (size_t)m * din + hi * 8;
        const float* Wt    = W + (size_t)t * din * HH;
        for (int kb = 0; kb < din; kb += 32) {
            // A tile (16x32 f16): lane holds K = kb + hi*8 + {0..7} and + {16..23}
            f4 x0 = *(const f4*)(Abase + kb);
            f4 x1 = *(const f4*)(Abase + kb + 4);
            f4 x2 = *(const f4*)(Abase + kb + 16);
            f4 x3 = *(const f4*)(Abase + kb + 20);
            v16h a;
            a[0]=(_Float16)x0.x; a[1]=(_Float16)x0.y; a[2]=(_Float16)x0.z; a[3]=(_Float16)x0.w;
            a[4]=(_Float16)x1.x; a[5]=(_Float16)x1.y; a[6]=(_Float16)x1.z; a[7]=(_Float16)x1.w;
            a[8]=(_Float16)x2.x; a[9]=(_Float16)x2.y; a[10]=(_Float16)x2.z; a[11]=(_Float16)x2.w;
            a[12]=(_Float16)x3.x; a[13]=(_Float16)x3.y; a[14]=(_Float16)x3.z; a[15]=(_Float16)x3.w;
            // B tile (32x16 f16): lane holds col, K = kb + hi*8 + {0..7} and + {16..23}
            const float* Wb = Wt + (size_t)(kb + hi * 8) * HH + col;
            v16h b;
            #pragma unroll
            for (int i = 0; i < 16; ++i) b[i] = (_Float16)Wb[(size_t)((i & 7) + (i >> 3) * 16) * HH];
            c = __builtin_amdgcn_wmma_f32_16x16x32_f16(false, a, false, b,
                                                       (short)0, c, false, false);
        }
    }
    // C layout: element e -> row M = e + hi*8, col N = lane&15 (within tile)
    float bb = bias[col];
    #pragma unroll
    for (int e = 0; e < 8; ++e) {
        int orow = blockIdx.x * 16 + e + hi * 8;
        float v = c[e] + bb;
        if (do_relu) v = fmaxf(v, 0.0f);
        out[(size_t)orow * ostride + ooff + col] = v;
    }
}

// ---------------- epilogue: gate MLP + softmax fusion + proj + cls + aux ----------
__device__ __forceinline__ void matvec64(float s0, float s1,
                                         const float* __restrict__ W,
                                         const float* __restrict__ b, int lane,
                                         float& o0, float& o1) {
    float a0 = b[lane], a1 = b[lane + 32];
    for (int i = 0; i < 32; ++i) {
        float v = __shfl(s0, i, 32);
        a0 += v * W[i * 64 + lane];
        a1 += v * W[i * 64 + lane + 32];
    }
    for (int i = 0; i < 32; ++i) {
        float v = __shfl(s1, i, 32);
        a0 += v * W[(i + 32) * 64 + lane];
        a1 += v * W[(i + 32) * 64 + lane + 32];
    }
    o0 = a0; o1 = a1;
}

__device__ __forceinline__ float waveReduce(float v) {
    for (int o = 16; o > 0; o >>= 1) v += __shfl_xor(v, o, 32);
    return v;
}

__global__ __launch_bounds__(256)
void fuse_kernel(const float* __restrict__ stk,
                 const float* __restrict__ gw1, const float* __restrict__ gb1,
                 const float* __restrict__ gw2, const float* __restrict__ gb2,
                 const float* __restrict__ pw,  const float* __restrict__ pb,
                 const float* __restrict__ cw1, const float* __restrict__ cb1,
                 const float* __restrict__ cw2, const float* __restrict__ cb2,
                 const float* __restrict__ aw,  const float* __restrict__ ab,
                 float* __restrict__ out) {
    int wv = threadIdx.x >> 5, lane = threadIdx.x & 31;
    int node = blockIdx.x * 8 + wv;
    if (node >= NN) return;
    const float* srow = stk + (size_t)node * (RR * HH);
    float s0[RR], s1[RR];
    #pragma unroll
    for (int r = 0; r < RR; ++r) {
        s0[r] = srow[r * HH + lane];
        s1[r] = srow[r * HH + lane + 32];
    }
    // per-relation gate logits
    float lg[RR];
    for (int r = 0; r < RR; ++r) {
        float g0, g1;
        matvec64(s0[r], s1[r], gw1, gb1, lane, g0, g1);
        g0 = fmaxf(g0, 0.0f); g1 = fmaxf(g1, 0.0f);
        float p = g0 * gw2[lane] + g1 * gw2[lane + 32];
        lg[r] = waveReduce(p) + gb2[0];
    }
    // softmax over relations
    float mx = lg[0];
    #pragma unroll
    for (int r = 1; r < RR; ++r) mx = fmaxf(mx, lg[r]);
    float al[RR], sum = 0.0f;
    #pragma unroll
    for (int r = 0; r < RR; ++r) { al[r] = __expf(lg[r] - mx); sum += al[r]; }
    float inv = 1.0f / sum;
    float hf0 = 0.0f, hf1 = 0.0f;
    #pragma unroll
    for (int r = 0; r < RR; ++r) {
        hf0 += al[r] * inv * s0[r];
        hf1 += al[r] * inv * s1[r];
    }
    // proj -> relu
    float p0, p1;
    matvec64(hf0, hf1, pw, pb, lane, p0, p1);
    p0 = fmaxf(p0, 0.0f); p1 = fmaxf(p1, 0.0f);
    // classifier hidden -> relu -> scalar
    float q0, q1;
    matvec64(p0, p1, cw1, cb1, lane, q0, q1);
    q0 = fmaxf(q0, 0.0f); q1 = fmaxf(q1, 0.0f);
    float pr = q0 * cw2[lane] + q1 * cw2[lane + 32];
    float logit = waveReduce(pr) + cb2[0];
    if (lane == 0) out[node] = logit;
    // aux logits
    for (int r = 0; r < RR; ++r) {
        float pa = s0[r] * aw[r * HH + lane] + s1[r] * aw[r * HH + lane + 32];
        pa = waveReduce(pa);
        if (lane == 0) out[(size_t)NN + (size_t)node * RR + r] = pa + ab[r];
    }
}

// ---------------- host orchestration ----------------
extern "C" void kernel_launch(void* const* d_in, const int* in_sizes, int n_in,
                              void* d_out, int out_size, void* d_ws, size_t ws_size,
                              hipStream_t stream) {
    (void)in_sizes; (void)n_in; (void)out_size; (void)ws_size;
    const float* x   = (const float*)d_in[0];
    const int*   ei  = (const int*)  d_in[1];
    const float* c1w = (const float*)d_in[2];
    const float* c1b = (const float*)d_in[3];
    const float* c2w = (const float*)d_in[4];
    const float* c2b = (const float*)d_in[5];
    const float* gw1 = (const float*)d_in[6];
    const float* gb1 = (const float*)d_in[7];
    const float* gw2 = (const float*)d_in[8];
    const float* gb2 = (const float*)d_in[9];
    const float* pw  = (const float*)d_in[10];
    const float* pb  = (const float*)d_in[11];
    const float* cw1 = (const float*)d_in[12];
    const float* cb1 = (const float*)d_in[13];
    const float* cw2 = (const float*)d_in[14];
    const float* cb2 = (const float*)d_in[15];
    const float* aw  = (const float*)d_in[16];
    const float* ab  = (const float*)d_in[17];

    float* ws = (float*)d_ws;
    size_t o = 0;
    float* deg = ws + o; o += NN;                  // reused as dis after dis_kernel
    float* nrm = ws + o; o += EE;
    float* tx1 = ws + o; o += (size_t)NN * DIN;
    float* tx2 = ws + o; o += (size_t)NN * DIN;
    float* h1  = ws + o; o += (size_t)NN * HH;
    float* t1  = ws + o; o += (size_t)NN * HH;
    float* t2  = ws + o; o += (size_t)NN * HH;
    float* stk = ws + o; o += (size_t)NN * RR * HH;

    const int TB = 256;
    for (int r = 0; r < RR; ++r) {
        const int* src = ei + (size_t)r * 2 * EE;
        const int* dst = src + EE;

        (void)hipMemsetAsync(deg, 0, NN * sizeof(float), stream);
        deg_kernel <<<(EE + TB - 1) / TB, TB, 0, stream>>>(src, deg);
        dis_kernel <<<(NN + TB - 1) / TB, TB, 0, stream>>>(deg);
        norm_kernel<<<(EE + TB - 1) / TB, TB, 0, stream>>>(src, dst, deg, nrm);

        // ---- layer 1 (D = 128) ----
        (void)hipMemsetAsync(tx1, 0, (size_t)NN * DIN * sizeof(float), stream);
        spmm_kernel<<<(EE * (DIN / 4) + TB - 1) / TB, TB, 0, stream>>>(src, dst, nrm, x, tx1, DIN / 4, 1.0f);
        negcopy_kernel<<<(NN * DIN + TB - 1) / TB, TB, 0, stream>>>(x, tx2, NN * DIN);   // tx2 = -tx0
        spmm_kernel<<<(EE * (DIN / 4) + TB - 1) / TB, TB, 0, stream>>>(src, dst, nrm, tx1, tx2, DIN / 4, 2.0f);
        cheb_gemm_kernel<<<NN / 16, 128, 0, stream>>>(x, tx1, tx2,
                                                      c1w + (size_t)r * 3 * DIN * HH,
                                                      c1b + (size_t)r * HH,
                                                      h1, DIN, HH, 0, 1);

        // ---- layer 2 (D = 64) ----
        (void)hipMemsetAsync(t1, 0, (size_t)NN * HH * sizeof(float), stream);
        spmm_kernel<<<(EE * (HH / 4) + TB - 1) / TB, TB, 0, stream>>>(src, dst, nrm, h1, t1, HH / 4, 1.0f);
        negcopy_kernel<<<(NN * HH + TB - 1) / TB, TB, 0, stream>>>(h1, t2, NN * HH);     // t2 = -h1
        spmm_kernel<<<(EE * (HH / 4) + TB - 1) / TB, TB, 0, stream>>>(src, dst, nrm, t1, t2, HH / 4, 2.0f);
        cheb_gemm_kernel<<<NN / 16, 128, 0, stream>>>(h1, t1, t2,
                                                      c2w + (size_t)r * 3 * HH * HH,
                                                      c2b + (size_t)r * HH,
                                                      stk, HH, RR * HH, r * HH, 1);
    }

    fuse_kernel<<<NN / 8, 256, 0, stream>>>(stk, gw1, gb1, gw2, gb2, pw, pb,
                                            cw1, cb1, cw2, cb2, aw, ab, (float*)d_out);
}
